// BigBirdRegressorHF_42150809043588
// MI455X (gfx1250) — compile-verified, matrix operand order checked by reading, not run on hardware
//
#include <hip/hip_runtime.h>
#include <math.h>

#define DEV static __device__ __forceinline__

typedef __bf16 bf16;
typedef bf16  v16bf __attribute__((ext_vector_type(16)));
typedef float v8f   __attribute__((ext_vector_type(8)));
typedef unsigned int u32x4 __attribute__((ext_vector_type(4)));
typedef int i32x4 __attribute__((ext_vector_type(4)));
typedef int i32x8 __attribute__((ext_vector_type(8)));

constexpr int Bc = 4, Sc = 4096, Dc = 768, Hc = 12, Lc = 4, BLKc = 64, DHc = 64;
constexpr int Mc = Bc * Sc;            // 16384 tokens
constexpr int FFc = 4 * Dc;            // 3072
constexpr int NBLK = Sc / BLKc;        // 64
constexpr int NI = NBLK - 2;           // 62 inner blocks
constexpr int NSc = 8;                 // kv blocks per inner block
constexpr float SCALEc = 0.125f;       // 1/sqrt(64)

constexpr size_t DDe   = (size_t)Dc * Dc;            // 589824
constexpr size_t DFe   = (size_t)Dc * FFc;           // 2359296
constexpr size_t LAYERe = 4 * DDe + 2 * DFe;         // packed elems / layer

// GEMM TDM staging: 4 k-steps (4KB) per chunk, double buffered, per wave.
constexpr int CHUNK = 4;
constexpr int GEMM_LDS = 8 /*waves*/ * 2 /*bufs*/ * CHUNK * 1024;  // 64KB

DEV v8f wmma_bf16(v16bf a, v16bf b, v8f c) {
  return __builtin_amdgcn_wmma_f32_16x16x32_bf16(false, a, false, b, (short)0, c, false, false);
}

// ---------------- Tensor Data Mover: 1-D contiguous panel -> LDS ----------------
// D# per CDNA5 ISA ch.8: group0 = {count/lds_addr/global_addr/type},
// group1 = {data_size, tensor/tile dims, strides}. 1 row of `ndw` dwords.
#if __has_builtin(__builtin_amdgcn_tensor_load_to_lds)
#define HAVE_TDM 1
DEV void tdm_load_1d(unsigned lds_byte_addr, const void* gaddr, int ndw) {
  unsigned long long ga = (unsigned long long)(uintptr_t)gaddr;
  u32x4 g0;
  g0.x = 1u;                                             // count=1, user D#
  g0.y = lds_byte_addr;                                  // lds_addr [63:32]
  g0.z = (unsigned)(ga & 0xffffffffull);                 // global_addr lo
  g0.w = (unsigned)((ga >> 32) & 0x1ffffffull) | (2u << 30);  // addr hi | type=2
  i32x8 g1;
  g1[0] = (2 << 16);                    // data_size=4B; no multicast/pad/iterate
  g1[1] = (ndw & 0xffff) << 16;         // tensor_dim0 lo16 (atomic_addr=0)
  g1[2] = (ndw >> 16) | (1 << 16);      // tensor_dim0 hi16 | tensor_dim1=1
  g1[3] = (ndw & 0xffff) << 16;         // tensor_dim1 hi16(0) | tile_dim0=ndw
  g1[4] = 0;                            // tile_dim1=0, tile_dim2=0 (1-D)
  g1[5] = ndw;                          // tensor_dim0_stride lo32
  g1[6] = 0;
  g1[7] = 0;
  i32x4 z4 = {0, 0, 0, 0};
#if __clang_major__ >= 23
  i32x8 z8 = {0, 0, 0, 0, 0, 0, 0, 0};
  __builtin_amdgcn_tensor_load_to_lds(g0, g1, z4, z4, z8, 0);
#else
  __builtin_amdgcn_tensor_load_to_lds(g0, g1, z4, z4, 0);
#endif
}
#else
#define HAVE_TDM 0
#endif

// A (or B) fragment gather from row-major [rows, ld] bf16.
// lane L holds row (L&15); K offsets: k0 + ((p&4)<<2) + ((L>>4)<<3) + ((p&3)<<1) + w
DEV v16bf frag_rowmajor(const bf16* base, size_t ld, int row, int k0, int lane) {
  v16bf f;
  const bf16* rp = base + (size_t)row * ld + k0 + ((lane >> 4) << 3);
#pragma unroll
  for (int p = 0; p < 8; p++) {
    int k = ((p & 4) << 2) + ((p & 3) << 1);
    f[2 * p]     = rp[k];
    f[2 * p + 1] = rp[k + 1];
  }
  return f;
}

// B fragment where the K dimension runs down rows of a row-major [s, ld] matrix
// (used for P·V: B[k=s, n=col] = V[s, col]). Strided gather.
DEV v16bf frag_colmajor(const bf16* base, size_t ld, int col, int s0, int lane) {
  v16bf f;
  const bf16* cp = base + col + (size_t)(s0 + ((lane >> 4) << 3)) * ld;
#pragma unroll
  for (int p = 0; p < 8; p++) {
    int k = ((p & 4) << 2) + ((p & 3) << 1);
    f[2 * p]     = cp[(size_t)k * ld];
    f[2 * p + 1] = cp[(size_t)(k + 1) * ld];
  }
  return f;
}

DEV float gelu_tanh(float x) {
  return 0.5f * x * (1.f + tanhf(0.7978845608f * (x + 0.044715f * x * x * x)));
}

// ---------------- weight packing: fp32 [K,N] -> bf16 B-fragment order -----------
__global__ __launch_bounds__(256) void k_pack(const float* __restrict__ W, bf16* __restrict__ out,
                                              int K, int N) {
  int idx = blockIdx.x * blockDim.x + threadIdx.x;
  int total = K * N;
  if (idx >= total) return;
  int e = idx & 15, lane = (idx >> 4) & 31, tile = idx >> 9;
  int ksteps = K >> 5;
  int nt = tile / ksteps, ks = tile % ksteps;
  int p = e >> 1, w = e & 1;
  int k = ks * 32 + ((p & 4) << 2) + ((lane >> 4) << 3) + ((p & 3) << 1) + w;
  int n = nt * 16 + (lane & 15);
  out[idx] = (bf16)W[(size_t)k * N + n];
}

// ---------------- bf16 WMMA GEMM: C[M,N] = A[M,K] @ Wp + bias (opt GELU) --------
// B panels are DMA'd to LDS by the Tensor Data Mover (double buffered per wave),
// A fragments stream directly from global (coalesces to global_load_b128).
__global__ __launch_bounds__(256) void k_gemm(const bf16* __restrict__ A,
                                              const bf16* __restrict__ Wp,
                                              const float* __restrict__ bias,
                                              float* Cf, bf16* Cb,
                                              int Mtiles, int Ntiles, int K, int act) {
  extern __shared__ char smem[];
  int wid  = (blockIdx.x * blockDim.x + threadIdx.x) >> 5;
  int lane = threadIdx.x & 31;
  int wv   = (threadIdx.x >> 5) & 7;
  if (wid >= Mtiles * Ntiles) return;
  int nt = wid % Ntiles, mt = wid / Ntiles;
  int N = Ntiles * 16, ks = K >> 5;
  const bf16* wpt = Wp + (size_t)nt * ks * 512;   // wave-uniform packed B panel
  int row = mt * 16 + (lane & 15);
  v8f acc = {};

#if HAVE_TDM
  bf16* lbuf = (bf16*)(smem + wv * 2 * CHUNK * 1024);
  unsigned lbase = (unsigned)(uintptr_t)lbuf;     // LDS byte offset (aperture low bits)
  int nch = ks / CHUNK;
  tdm_load_1d(lbase, wpt, CHUNK * 256);           // prologue: chunk 0
  for (int c = 0; c < nch; c++) {
    if (c + 1 < nch) {
      tdm_load_1d(lbase + (unsigned)(((c + 1) & 1) * CHUNK * 1024),
                  wpt + (size_t)(c + 1) * CHUNK * 512, CHUNK * 256);
      __builtin_amdgcn_s_wait_tensorcnt((short)1);   // chunk c has landed
    } else {
      __builtin_amdgcn_s_wait_tensorcnt((short)0);
    }
    const bf16* cb = lbuf + (size_t)(c & 1) * CHUNK * 512 + lane * 16;
#pragma unroll
    for (int s = 0; s < CHUNK; s++) {
      v16bf a = frag_rowmajor(A, (size_t)K, row, (c * CHUNK + s) * 32, lane);
      v16bf b = *(const v16bf*)(cb + (size_t)s * 512);
      acc = wmma_bf16(a, b, acc);
    }
  }
#else
  const bf16* wl = wpt + lane * 16;
  for (int s = 0; s < ks; s++) {
    v16bf a = frag_rowmajor(A, (size_t)K, row, s * 32, lane);
    v16bf b = *(const v16bf*)(wl + (size_t)s * 512);
    if (s + 1 < ks) __builtin_prefetch(wl + (size_t)(s + 1) * 512, 0, 1);
    acc = wmma_bf16(a, b, acc);
  }
#endif

  int col = nt * 16 + (lane & 15);
  float bv = bias ? bias[col] : 0.f;
  int hi = (lane >> 4) << 3;
#pragma unroll
  for (int r = 0; r < 8; r++) {
    float v = acc[r] + bv;
    if (act) v = gelu_tanh(v);
    size_t o = (size_t)(mt * 16 + r + hi) * N + col;
    if (Cf) Cf[o] = v;
    if (Cb) Cb[o] = (bf16)v;
  }
}

// ---------------- embedding + LayerNorm ----------------------------------------
__global__ __launch_bounds__(256) void k_embed(const float* __restrict__ in,
                                               const float* __restrict__ pos,
                                               const float* __restrict__ tok,
                                               const float* __restrict__ lnw,
                                               float* __restrict__ h, bf16* __restrict__ hb) {
  int row = blockIdx.x, t = threadIdx.x;
  __shared__ float red[256];
  float v[3]; float s = 0.f;
#pragma unroll
  for (int j = 0; j < 3; j++) {
    int c = t + j * 256;
    float x = in[(size_t)row * Dc + c] + pos[(size_t)(row % Sc) * Dc + c] + tok[c];
    v[j] = x; s += x;
  }
  red[t] = s; __syncthreads();
  for (int o = 128; o > 0; o >>= 1) { if (t < o) red[t] += red[t + o]; __syncthreads(); }
  float mu = red[0] / (float)Dc; __syncthreads();
  float vs = 0.f;
#pragma unroll
  for (int j = 0; j < 3; j++) { float d = v[j] - mu; vs += d * d; }
  red[t] = vs; __syncthreads();
  for (int o = 128; o > 0; o >>= 1) { if (t < o) red[t] += red[t + o]; __syncthreads(); }
  float rstd = rsqrtf(red[0] / (float)Dc + 1e-12f);
#pragma unroll
  for (int j = 0; j < 3; j++) {
    int c = t + j * 256;
    float y = (v[j] - mu) * rstd * lnw[c] + lnw[Dc + c];
    h[(size_t)row * Dc + c] = y;
    hb[(size_t)row * Dc + c] = (bf16)y;
  }
}

// ---------------- residual + LayerNorm (h = LN(h + x)) -------------------------
__global__ __launch_bounds__(256) void k_resln(float* __restrict__ h, const float* __restrict__ x,
                                               const float* __restrict__ lnw, bf16* __restrict__ hb) {
  int row = blockIdx.x, t = threadIdx.x;
  __shared__ float red[256];
  float v[3]; float s = 0.f;
#pragma unroll
  for (int j = 0; j < 3; j++) {
    int c = t + j * 256;
    float val = h[(size_t)row * Dc + c] + x[(size_t)row * Dc + c];
    v[j] = val; s += val;
  }
  red[t] = s; __syncthreads();
  for (int o = 128; o > 0; o >>= 1) { if (t < o) red[t] += red[t + o]; __syncthreads(); }
  float mu = red[0] / (float)Dc; __syncthreads();
  float vs = 0.f;
#pragma unroll
  for (int j = 0; j < 3; j++) { float d = v[j] - mu; vs += d * d; }
  red[t] = vs; __syncthreads();
  for (int o = 128; o > 0; o >>= 1) { if (t < o) red[t] += red[t + o]; __syncthreads(); }
  float rstd = rsqrtf(red[0] / (float)Dc + 1e-12f);
#pragma unroll
  for (int j = 0; j < 3; j++) {
    int c = t + j * 256;
    float y = (v[j] - mu) * rstd * lnw[c] + lnw[Dc + c];
    h[(size_t)row * Dc + c] = y;
    hb[(size_t)row * Dc + c] = (bf16)y;
  }
}

// ---------------- inner block-sparse attention ---------------------------------
// grid (NI, H, B), block 128 (4 waves). dyn-LDS: 64*512 bf16 = 64KB
__global__ __launch_bounds__(128) void k_attn_inner(const bf16* __restrict__ Q,
                                                    const bf16* __restrict__ Kx,
                                                    const bf16* __restrict__ Vx,
                                                    bf16* __restrict__ ctx) {
  extern __shared__ char smem[];
  bf16* Sp = (bf16*)smem;  // [64][512] scores then probabilities
  __shared__ int   kvb[NSc];
  __shared__ float kvm[NSc];
  int i = blockIdx.x + 1, h = blockIdx.y, b = blockIdx.z;
  int lane = threadIdx.x & 31, wave = threadIdx.x >> 5;

  if (threadIdx.x == 0) {
    int sel[NSc];
    sel[0] = 0; sel[1] = NBLK - 1; sel[2] = i - 1; sel[3] = i; sel[4] = i + 1;
    unsigned st = 42u ^ (2654435761u * (unsigned)i);
    for (int r = 5; r < NSc; r++) {
      st = st * 1664525u + 1013904223u;
      sel[r] = 1 + (int)(st % (unsigned)(NBLK - 2));   // 1..62 (LCG stand-in for np RNG)
    }
    for (int g = 0; g < NSc; g++) {
      kvb[g] = sel[g];
      bool dup = false;
      for (int p = 0; p < g; p++) dup |= (sel[p] == sel[g]);
      kvm[g] = dup ? 0.f : 1.f;                        // first-occurrence mask
    }
  }
  __syncthreads();

  const bf16* qb = Q  + (size_t)b * Sc * Dc + h * DHc;
  const bf16* kb = Kx + (size_t)b * Sc * Dc + h * DHc;
  const bf16* vb = Vx + (size_t)b * Sc * Dc;
  int hi = (lane >> 4) << 3;
  int qrow = i * BLKc + wave * 16 + (lane & 15);
  v16bf a0 = frag_rowmajor(qb, Dc, qrow, 0, lane);
  v16bf a1 = frag_rowmajor(qb, Dc, qrow, 32, lane);

  // scores S = (Q K^T) * scale, masked, staged bf16 in LDS
  for (int g = 0; g < NSc; g++) {
    float msk = kvm[g];
    for (int t = 0; t < 4; t++) {
      int krow = kvb[g] * BLKc + t * 16 + (lane & 15);
      v16bf b0 = frag_rowmajor(kb, Dc, krow, 0, lane);
      v16bf b1 = frag_rowmajor(kb, Dc, krow, 32, lane);
      v8f acc = {};
      acc = wmma_bf16(a0, b0, acc);
      acc = wmma_bf16(a1, b1, acc);
#pragma unroll
      for (int r = 0; r < 8; r++) {
        float sv = (msk != 0.f) ? acc[r] * SCALEc : -1.0e9f;
        Sp[(size_t)(wave * 16 + r + hi) * 512 + g * 64 + t * 16 + (lane & 15)] = (bf16)sv;
      }
    }
  }
  __syncthreads();

  // row softmax (one thread per query row)
  if (threadIdx.x < 64) {
    bf16* rp = Sp + (size_t)threadIdx.x * 512;
    float mx = -3.0e38f;
    for (int c = 0; c < 512; c++) { float v = (float)rp[c]; mx = v > mx ? v : mx; }
    float sm = 0.f;
    for (int c = 0; c < 512; c++) { float e = __expf((float)rp[c] - mx); sm += e; rp[c] = (bf16)e; }
    float inv = 1.f / sm;
    for (int c = 0; c < 512; c++) rp[c] = (bf16)((float)rp[c] * inv);
  }
  __syncthreads();

  // ctx = P @ V_gathered
  v8f o[4] = {};
  int prow = wave * 16 + (lane & 15);
  for (int s = 0; s < 16; s++) {
    v16bf pa;
    const bf16* pp = Sp + (size_t)prow * 512 + s * 32 + ((lane >> 4) << 3);
#pragma unroll
    for (int p = 0; p < 8; p++) {
      int k = ((p & 4) << 2) + ((p & 3) << 1);
      pa[2 * p] = pp[k]; pa[2 * p + 1] = pp[k + 1];
    }
    int g = s >> 1;
    int s0 = kvb[g] * BLKc + (s & 1) * 32;
#pragma unroll
    for (int n = 0; n < 4; n++) {
      v16bf bv = frag_colmajor(vb, Dc, h * DHc + n * 16 + (lane & 15), s0, lane);
      o[n] = wmma_bf16(pa, bv, o[n]);
    }
  }
#pragma unroll
  for (int n = 0; n < 4; n++)
#pragma unroll
    for (int r = 0; r < 8; r++) {
      size_t tk = (size_t)b * Sc + i * BLKc + wave * 16 + r + hi;
      ctx[tk * Dc + h * DHc + n * 16 + (lane & 15)] = (bf16)o[n][r];
    }
}

// ---------------- edge (global) attention: flash-style over full S -------------
// grid (2, H, B), block 128 (4 waves)
__global__ __launch_bounds__(128) void k_attn_edge(const bf16* __restrict__ Q,
                                                   const bf16* __restrict__ Kx,
                                                   const bf16* __restrict__ Vx,
                                                   bf16* __restrict__ ctx) {
  __shared__ bf16 Pst[4][16 * 32];
  int e = blockIdx.x, h = blockIdx.y, b = blockIdx.z;
  int lane = threadIdx.x & 31, wave = threadIdx.x >> 5;
  const bf16* qb = Q  + (size_t)b * Sc * Dc + h * DHc;
  const bf16* kb = Kx + (size_t)b * Sc * Dc + h * DHc;
  const bf16* vb = Vx + (size_t)b * Sc * Dc;
  int hi = (lane >> 4) << 3;
  int qbase = (e == 0 ? 0 : NBLK - 1) * BLKc;
  int qrow = qbase + wave * 16 + (lane & 15);
  v16bf a0 = frag_rowmajor(qb, Dc, qrow, 0, lane);
  v16bf a1 = frag_rowmajor(qb, Dc, qrow, 32, lane);

  float m[8], l[8];
  v8f o[4] = {};
#pragma unroll
  for (int r = 0; r < 8; r++) { m[r] = -3.0e38f; l[r] = 0.f; }

  for (int step = 0; step < Sc / 32; step++) {
    v8f st[2];
#pragma unroll
    for (int t = 0; t < 2; t++) {
      int krow = step * 32 + t * 16 + (lane & 15);
      v16bf b0 = frag_rowmajor(kb, Dc, krow, 0, lane);
      v16bf b1 = frag_rowmajor(kb, Dc, krow, 32, lane);
      v8f acc = {};
      acc = wmma_bf16(a0, b0, acc);
      acc = wmma_bf16(a1, b1, acc);
      st[t] = acc;
    }
    float e0[8], e1[8];
#pragma unroll
    for (int r = 0; r < 8; r++) {
      float s0 = st[0][r] * SCALEc, s1 = st[1][r] * SCALEc;
      float tm = fmaxf(s0, s1);
      for (int d = 1; d < 16; d <<= 1) tm = fmaxf(tm, __shfl_xor(tm, d, 32));
      float nm = fmaxf(m[r], tm);
      float c  = __expf(m[r] - nm);
      float p0 = __expf(s0 - nm), p1 = __expf(s1 - nm);
      float rs = p0 + p1;
      for (int d = 1; d < 16; d <<= 1) rs += __shfl_xor(rs, d, 32);
      l[r] = l[r] * c + rs;
      m[r] = nm;
      e0[r] = p0; e1[r] = p1;
#pragma unroll
      for (int n = 0; n < 4; n++) o[n][r] *= c;
    }
    __syncthreads();
#pragma unroll
    for (int r = 0; r < 8; r++) {
      Pst[wave][(r + hi) * 32 + (lane & 15)]      = (bf16)e0[r];
      Pst[wave][(r + hi) * 32 + 16 + (lane & 15)] = (bf16)e1[r];
    }
    __syncthreads();
    v16bf pa;
    const bf16* pp = &Pst[wave][(lane & 15) * 32 + ((lane >> 4) << 3)];
#pragma unroll
    for (int p = 0; p < 8; p++) {
      int k = ((p & 4) << 2) + ((p & 3) << 1);
      pa[2 * p] = pp[k]; pa[2 * p + 1] = pp[k + 1];
    }
#pragma unroll
    for (int n = 0; n < 4; n++) {
      v16bf bv = frag_colmajor(vb, Dc, h * DHc + n * 16 + (lane & 15), step * 32, lane);
      o[n] = wmma_bf16(pa, bv, o[n]);
    }
  }
#pragma unroll
  for (int n = 0; n < 4; n++)
#pragma unroll
    for (int r = 0; r < 8; r++) {
      size_t tk = (size_t)b * Sc + qbase + wave * 16 + r + hi;
      ctx[tk * Dc + h * DHc + n * 16 + (lane & 15)] = (bf16)(o[n][r] / l[r]);
    }
}

// ---------------- pooling + head ------------------------------------------------
__global__ __launch_bounds__(256) void k_pool(const float* __restrict__ h, float* __restrict__ pooled) {
  int idx = blockIdx.x * blockDim.x + threadIdx.x;
  if (idx >= Bc * Dc) return;
  int b = idx / Dc, d = idx % Dc;
  float s = 0.f;
  for (int t = 0; t < Sc; t++) s += h[((size_t)b * Sc + t) * Dc + d];
  pooled[idx] = s / (float)Sc;
}
__global__ __launch_bounds__(256) void k_dense(const float* __restrict__ pooled,
                                               const float* __restrict__ w,
                                               const float* __restrict__ bi,
                                               float* __restrict__ tv) {
  int idx = blockIdx.x * blockDim.x + threadIdx.x;
  if (idx >= Bc * Dc) return;
  int b = idx / Dc, e2 = idx % Dc;
  float s = bi[e2];
  for (int d = 0; d < Dc; d++) s += pooled[b * Dc + d] * w[(size_t)d * Dc + e2];
  tv[idx] = tanhf(s);
}
__global__ void k_logits(const float* __restrict__ tv, const float* __restrict__ w,
                         const float* __restrict__ bi, float* __restrict__ out) {
  int b = threadIdx.x;
  if (b >= Bc) return;
  float s = bi[0];
  for (int d = 0; d < Dc; d++) s += tv[b * Dc + d] * w[d];
  out[b] = s;
}

// ---------------- host orchestration -------------------------------------------
extern "C" void kernel_launch(void* const* d_in, const int* in_sizes, int n_in,
                              void* d_out, int out_size, void* d_ws, size_t ws_size,
                              hipStream_t stream) {
  (void)in_sizes; (void)n_in; (void)out_size; (void)ws_size;
  const float* in_emb = (const float*)d_in[0];
  const float* pos    = (const float*)d_in[1];
  const float* tok    = (const float*)d_in[2];
  const float* embln  = (const float*)d_in[3];
  const float* qkvo_w = (const float*)d_in[4];
  const float* qkvo_b = (const float*)d_in[5];
  const float* ln1    = (const float*)d_in[6];
  const float* ln2    = (const float*)d_in[7];
  const float* ffn_wi = (const float*)d_in[8];
  const float* ffn_bi = (const float*)d_in[9];
  const float* ffn_wo = (const float*)d_in[10];
  const float* ffn_bo = (const float*)d_in[11];
  const float* dw     = (const float*)d_in[12];
  const float* db     = (const float*)d_in[13];
  const float* ow     = (const float*)d_in[14];
  const float* ob     = (const float*)d_in[15];
  float* out = (float*)d_out;

  char* ws = (char*)d_ws;
  size_t off = 0;
  auto alloc = [&](size_t bytes) -> void* {
    void* p = ws + off; off += (bytes + 255) & ~(size_t)255; return p;
  };
  float* h    = (float*)alloc((size_t)Mc * Dc * 4);
  float* tmp  = (float*)alloc((size_t)Mc * Dc * 4);
  bf16*  hb   = (bf16*) alloc((size_t)Mc * Dc * 2);
  bf16*  qbuf = (bf16*) alloc((size_t)Mc * Dc * 2);
  bf16*  kbuf = (bf16*) alloc((size_t)Mc * Dc * 2);
  bf16*  vbuf = (bf16*) alloc((size_t)Mc * Dc * 2);
  bf16*  ctxb = (bf16*) alloc((size_t)Mc * Dc * 2);
  bf16*  f1b  = (bf16*) alloc((size_t)Mc * FFc * 2);
  bf16*  wp   = (bf16*) alloc(LAYERe * Lc * 2);
  float* pooled = (float*)alloc((size_t)Bc * Dc * 4);
  float* tv     = (float*)alloc((size_t)Bc * Dc * 4);

  auto pack = [&](const float* W, bf16* o, int K, int N) {
    int total = K * N;
    k_pack<<<(total + 255) / 256, 256, 0, stream>>>(W, o, K, N);
  };
  auto gemm = [&](const bf16* A, const bf16* Wpk, const float* bias,
                  float* Cf, bf16* Cb, int Mt, int Nt, int K, int act) {
    long long waves = (long long)Mt * Nt;
    long long blocks = (waves * 32 + 255) / 256;
    k_gemm<<<(unsigned)blocks, 256, GEMM_LDS, stream>>>(A, Wpk, bias, Cf, Cb, Mt, Nt, K, act);
  };

  // 1) pack all weights to bf16 fragment layout
  for (int l = 0; l < Lc; l++) {
    bf16* lw = wp + (size_t)l * LAYERe;
    for (int i2 = 0; i2 < 4; i2++)
      pack(qkvo_w + ((size_t)l * 4 + i2) * DDe, lw + (size_t)i2 * DDe, Dc, Dc);
    pack(ffn_wi + (size_t)l * DFe, lw + 4 * DDe, Dc, FFc);
    pack(ffn_wo + (size_t)l * DFe, lw + 4 * DDe + DFe, FFc, Dc);
  }

  // 2) embeddings + LN
  k_embed<<<Mc, 256, 0, stream>>>(in_emb, pos, tok, embln, h, hb);

  // 3) layers
  for (int l = 0; l < Lc; l++) {
    bf16* lw = wp + (size_t)l * LAYERe;
    const float* qb_b = qkvo_b + ((size_t)l * 4 + 0) * Dc;
    const float* kb_b = qkvo_b + ((size_t)l * 4 + 1) * Dc;
    const float* vb_b = qkvo_b + ((size_t)l * 4 + 2) * Dc;
    const float* ob_b = qkvo_b + ((size_t)l * 4 + 3) * Dc;

    gemm(hb, lw + 0 * DDe, qb_b, nullptr, qbuf, Mc / 16, Dc / 16, Dc, 0);
    gemm(hb, lw + 1 * DDe, kb_b, nullptr, kbuf, Mc / 16, Dc / 16, Dc, 0);
    gemm(hb, lw + 2 * DDe, vb_b, nullptr, vbuf, Mc / 16, Dc / 16, Dc, 0);

    k_attn_inner<<<dim3(NI, Hc, Bc), 128, 64 * 512 * sizeof(bf16), stream>>>(qbuf, kbuf, vbuf, ctxb);
    k_attn_edge <<<dim3(2,  Hc, Bc), 128, 0, stream>>>(qbuf, kbuf, vbuf, ctxb);

    gemm(ctxb, lw + 3 * DDe, ob_b, tmp, nullptr, Mc / 16, Dc / 16, Dc, 0);
    k_resln<<<Mc, 256, 0, stream>>>(h, tmp, ln1 + (size_t)l * 2 * Dc, hb);

    gemm(hb,  lw + 4 * DDe,       ffn_bi + (size_t)l * FFc, nullptr, f1b, Mc / 16, FFc / 16, Dc, 1);
    gemm(f1b, lw + 4 * DDe + DFe, ffn_bo + (size_t)l * Dc,  tmp, nullptr, Mc / 16, Dc / 16, FFc, 0);
    k_resln<<<Mc, 256, 0, stream>>>(h, tmp, ln2 + (size_t)l * 2 * Dc, hb);
  }

  // 4) head
  k_pool  <<<(Bc * Dc + 255) / 256, 256, 0, stream>>>(h, pooled);
  k_dense <<<(Bc * Dc + 255) / 256, 256, 0, stream>>>(pooled, dw, db, tv);
  k_logits<<<1, 64, 0, stream>>>(tv, ow, ob, out);
}
